// DNN_sym_new_56822417326603
// MI455X (gfx1250) — matrix operand order repaired and checked by reading, not compiled, not run on hardware
//
#include <hip/hip_runtime.h>

typedef __attribute__((ext_vector_type(16))) _Float16 v16h;
typedef __attribute__((ext_vector_type(8)))  _Float16 v8h;
typedef __attribute__((ext_vector_type(8)))  float    v8f;

union V16U { v16h v; v8h h[2]; };

#define NWAVES 8   // 256 threads / wave32

// slope 0.01 < 1  =>  leaky_relu(v) == max(v, 0.01*v)
__device__ __forceinline__ float lrelu(float v) { return fmaxf(v, 0.01f * v); }

// ---------------------------------------------------------------------------
// Workspace layout (bytes):
//   0    : dpart[192]  float   partial aggregation
//   768  : cnt[4]      int     per-type histogram
//   784  : cur[4]      int     scatter cursors
//   800  : off[5]      int     16-aligned region starts, off[4] = padded total
//   1024 : x_sorted[3*(n+64)] float
// ---------------------------------------------------------------------------

__global__ void k_init(float* __restrict__ dpart, int* __restrict__ cnt) {
    int i = threadIdx.x;
    if (i < 192) dpart[i] = 0.0f;
    if (i < 4)   cnt[i] = 0;
}

__global__ void k_hist(const int* __restrict__ types, int* __restrict__ cnt, int n) {
    __shared__ int h[4];
    if (threadIdx.x < 4) h[threadIdx.x] = 0;
    __syncthreads();
    for (int i = blockIdx.x * blockDim.x + threadIdx.x; i < n; i += gridDim.x * blockDim.x)
        atomicAdd(&h[types[i]], 1);
    __syncthreads();
    if (threadIdx.x < 4) atomicAdd(&cnt[threadIdx.x], h[threadIdx.x]);
}

__global__ void k_offs(const int* __restrict__ cnt, int* __restrict__ cur,
                       int* __restrict__ off, float* __restrict__ xsorted) {
    __shared__ int so[5];
    int tid = threadIdx.x;
    if (tid == 0) {
        int o = 0;
        for (int t = 0; t < 4; ++t) { so[t] = o; o += (cnt[t] + 15) & ~15; }
        so[4] = o;
    }
    __syncthreads();
    if (tid < 4) { off[tid] = so[tid]; cur[tid] = so[tid]; }
    if (tid == 4) off[4] = so[4];
    // zero-fill pad slots (<=15 per type): zero x => zero contribution to d
    for (int t = 0; t < 4; ++t) {
        int s = cnt[t], e = (s + 15) & ~15;
        int i = s + tid;
        if (i < e) {
            int p = so[t] + i;
            xsorted[p * 3 + 0] = 0.0f;
            xsorted[p * 3 + 1] = 0.0f;
            xsorted[p * 3 + 2] = 0.0f;
        }
    }
}

__global__ void k_scatter(const float* __restrict__ x, const int* __restrict__ types,
                          int* __restrict__ cur, float* __restrict__ xsorted, int n) {
    int lane = threadIdx.x & 31;
    int gw = blockIdx.x * (blockDim.x >> 5) + (threadIdx.x >> 5);
    int sw = gridDim.x * (blockDim.x >> 5);
    for (int aw = gw; aw * 32 < n; aw += sw) {
        int a = aw * 32 + lane;
        bool valid = (a < n);
        int ty = -1;
        float x0 = 0.f, x1 = 0.f, x2 = 0.f;
        if (valid) {
            ty = types[a];
            x0 = x[a * 3 + 0]; x1 = x[a * 3 + 1]; x2 = x[a * 3 + 2];
        }
        for (int t = 0; t < 4; ++t) {
            unsigned m = (unsigned)__ballot(ty == t);
            if (m) {                                   // wave-uniform
                int cw = __popc(m);
                int leader = __ffs(m) - 1;
                int base = 0;
                if (lane == leader) base = atomicAdd(&cur[t], cw);
                base = __shfl(base, leader);
                if (ty == t) {
                    int p = base + __popc(m & ((1u << lane) - 1u));
                    xsorted[p * 3 + 0] = x0;
                    xsorted[p * 3 + 1] = x1;
                    xsorted[p * 3 + 2] = x2;
                }
            }
        }
    }
}

// ---------------------------------------------------------------------------
// Fast path: type-sorted atoms -> every 16-atom WMMA tile is type-uniform.
// 6 WMMAs per tile, no selection, activation fused straight into g^T x FMAs.
// ---------------------------------------------------------------------------
__global__ void __launch_bounds__(256)
k_embed_sorted(const float* __restrict__ xsorted,
               const float* __restrict__ W0, const float* __restrict__ b0,
               const float* __restrict__ W1, const float* __restrict__ b1,
               const int*   __restrict__ off,
               float* __restrict__ dpart) {
    __shared__ __align__(32) _Float16 sB0[4][2][32][16];  // K0..2 = W0, K3 = bias row
    __shared__ __align__(32) _Float16 sB1[4][4][32][16];
    __shared__ float sb1[4][64];
    __shared__ __align__(32) _Float16 h0s[NWAVES][16][32];
    __shared__ float xs[NWAVES][16][3];

    for (int slot = threadIdx.x; slot < 4 * 2 * 32; slot += blockDim.x) {
        int t = slot >> 6, ch = (slot >> 5) & 1, l = slot & 31;
        int nn = ch * 16 + (l & 15);
        _Float16* dst = sB0[t][ch][l];
        for (int i = 0; i < 16; ++i) dst[i] = (_Float16)0.0f;
        if (l < 16) {
            dst[0] = (_Float16)W0[(t * 3 + 0) * 32 + nn];
            dst[1] = (_Float16)W0[(t * 3 + 1) * 32 + nn];
            dst[2] = (_Float16)W0[(t * 3 + 2) * 32 + nn];
            dst[3] = (_Float16)b0[t * 32 + nn];
        }
    }
    for (int slot = threadIdx.x; slot < 4 * 4 * 32; slot += blockDim.x) {
        int t = slot >> 7, j = (slot >> 5) & 3, l = slot & 31;
        int nn = j * 16 + (l & 15);
        int koff = (l < 16) ? 0 : 16;
        _Float16* dst = sB1[t][j][l];
        for (int i = 0; i < 16; ++i)
            dst[i] = (_Float16)W1[(t * 32 + koff + i) * 64 + nn];
    }
    for (int i = threadIdx.x; i < 256; i += blockDim.x) sb1[i >> 6][i & 63] = b1[i];
    __syncthreads();

    const int wave  = threadIdx.x >> 5;
    const int lane  = threadIdx.x & 31;
    const int lh    = lane & 15;
    const bool lolane = (lane < 16);
    const int mbase = lolane ? 0 : 8;
    const int kb    = lolane ? 0 : 16;

    const int o1 = off[1], o2 = off[2], o3 = off[3];
    const int TT = off[4] >> 4;   // padded total is 16-aligned

    float dacc[4][3] = {};

    for (int tile = blockIdx.x * NWAVES + wave; tile < TT; tile += gridDim.x * NWAVES) {
        int base = tile * 16;
        int t = (base >= o1) + (base >= o2) + (base >= o3);  // region -> type

        float x0 = 0.f, x1 = 0.f, x2 = 0.f;
        if (lolane) {
            int a = base + lh;                  // always in-bounds (padded regions)
            x0 = xsorted[a * 3 + 0]; x1 = xsorted[a * 3 + 1]; x2 = xsorted[a * 3 + 2];
            xs[wave][lh][0] = x0; xs[wave][lh][1] = x1; xs[wave][lh][2] = x2;
        }
        __builtin_amdgcn_wave_barrier();

        v16h a0 = {};
        if (lolane) {
            a0[0] = (_Float16)x0; a0[1] = (_Float16)x1; a0[2] = (_Float16)x2;
            a0[3] = (_Float16)1.0f;             // bias row
        }

        // ---- layer 0 ----
        {
            v16h b0a = *(const v16h*)sB0[t][0][lane];
            v8f c0a = {};
            c0a = __builtin_amdgcn_wmma_f32_16x16x32_f16(false, a0, false, b0a,
                                                         (short)0, c0a, false, false);
            for (int r = 0; r < 8; ++r)
                h0s[wave][mbase + r][lh] = (_Float16)lrelu(c0a[r]);
        }
        {
            v16h b0b = *(const v16h*)sB0[t][1][lane];
            v8f c0b = {};
            c0b = __builtin_amdgcn_wmma_f32_16x16x32_f16(false, a0, false, b0b,
                                                         (short)0, c0b, false, false);
            for (int r = 0; r < 8; ++r)
                h0s[wave][mbase + r][16 + lh] = (_Float16)lrelu(c0b[r]);
        }
        __builtin_amdgcn_wave_barrier();

        // ---- transpose C-layout -> A-layout via LDS ----
        V16U a1;
        a1.h[0] = *(const v8h*)&h0s[wave][lh][kb];
        a1.h[1] = *(const v8h*)&h0s[wave][lh][kb + 8];
        __builtin_amdgcn_wave_barrier();

        // ---- layer 1: 4 independent WMMAs back-to-back ----
        v8f c1v[4];
        for (int j = 0; j < 4; ++j) {
            v16h b1f = *(const v16h*)sB1[t][j][lane];
            v8f c1 = {};
            c1v[j] = __builtin_amdgcn_wmma_f32_16x16x32_f16(false, a1.v, false, b1f,
                                                            (short)0, c1, false, false);
        }
        float bb[4];
        for (int j = 0; j < 4; ++j) bb[j] = sb1[t][j * 16 + lh];

        // ---- fused activation + partial aggregation d[e,c] += g[m,e]*x[m,c] ----
        for (int r = 0; r < 8; ++r) {
            int m = mbase + r;
            float xr0 = xs[wave][m][0], xr1 = xs[wave][m][1], xr2 = xs[wave][m][2];
            for (int j = 0; j < 4; ++j) {
                float gv = lrelu(c1v[j][r] + bb[j]);
                dacc[j][0] += gv * xr0;
                dacc[j][1] += gv * xr1;
                dacc[j][2] += gv * xr2;
            }
        }
    }

    for (int j = 0; j < 4; ++j)
        for (int c = 0; c < 3; ++c)
            unsafeAtomicAdd(&dpart[(j * 16 + lh) * 3 + c], dacc[j][c]);
}

// ---------------------------------------------------------------------------
// Fallback (workspace too small to sort): mixed-type tiles, 4 types + select.
// ---------------------------------------------------------------------------
__global__ void __launch_bounds__(256)
k_embed_mixed(const float* __restrict__ x, const int* __restrict__ types,
              const float* __restrict__ W0, const float* __restrict__ b0,
              const float* __restrict__ W1, const float* __restrict__ b1,
              float* __restrict__ dpart, int n) {
    __shared__ __align__(32) _Float16 sB0[4][2][32][16];
    __shared__ __align__(32) _Float16 sB1[4][4][32][16];
    __shared__ float sb1[4][64];
    __shared__ __align__(32) _Float16 h0s[NWAVES][16][32];
    __shared__ float xs[NWAVES][16][3];
    __shared__ __align__(16) int tss[NWAVES][16];

    for (int slot = threadIdx.x; slot < 4 * 2 * 32; slot += blockDim.x) {
        int t = slot >> 6, ch = (slot >> 5) & 1, l = slot & 31;
        int nn = ch * 16 + (l & 15);
        _Float16* dst = sB0[t][ch][l];
        for (int i = 0; i < 16; ++i) dst[i] = (_Float16)0.0f;
        if (l < 16) {
            dst[0] = (_Float16)W0[(t * 3 + 0) * 32 + nn];
            dst[1] = (_Float16)W0[(t * 3 + 1) * 32 + nn];
            dst[2] = (_Float16)W0[(t * 3 + 2) * 32 + nn];
            dst[3] = (_Float16)b0[t * 32 + nn];
        }
    }
    for (int slot = threadIdx.x; slot < 4 * 4 * 32; slot += blockDim.x) {
        int t = slot >> 7, j = (slot >> 5) & 3, l = slot & 31;
        int nn = j * 16 + (l & 15);
        int koff = (l < 16) ? 0 : 16;
        _Float16* dst = sB1[t][j][l];
        for (int i = 0; i < 16; ++i)
            dst[i] = (_Float16)W1[(t * 32 + koff + i) * 64 + nn];
    }
    for (int i = threadIdx.x; i < 256; i += blockDim.x) sb1[i >> 6][i & 63] = b1[i];
    __syncthreads();

    const int wave  = threadIdx.x >> 5;
    const int lane  = threadIdx.x & 31;
    const int lh    = lane & 15;
    const bool lolane = (lane < 16);
    const int mbase = lolane ? 0 : 8;
    const int kb    = lolane ? 0 : 16;
    const int TT    = (n + 15) >> 4;

    float dacc[4][3] = {};

    for (int tile = blockIdx.x * NWAVES + wave; tile < TT; tile += gridDim.x * NWAVES) {
        float x0 = 0.f, x1 = 0.f, x2 = 0.f;
        int ty = -1;
        if (lolane) {
            int a = tile * 16 + lh;
            if (a < n) {
                x0 = x[a * 3 + 0]; x1 = x[a * 3 + 1]; x2 = x[a * 3 + 2];
                ty = types[a];
            }
            xs[wave][lh][0] = x0; xs[wave][lh][1] = x1; xs[wave][lh][2] = x2;
            tss[wave][lh] = ty;
        }
        __builtin_amdgcn_wave_barrier();

        v16h a0 = {};
        if (lolane) {
            a0[0] = (_Float16)x0; a0[1] = (_Float16)x1; a0[2] = (_Float16)x2;
            a0[3] = (_Float16)1.0f;
        }
        int tsel[8];
        {
            const int* tp = &tss[wave][mbase];
            for (int r = 0; r < 8; ++r) tsel[r] = tp[r];
        }
        v8f gacc[4];

        for (int t = 0; t < 4; ++t) {
            {
                v16h b0a = *(const v16h*)sB0[t][0][lane];
                v8f c0a = {};
                c0a = __builtin_amdgcn_wmma_f32_16x16x32_f16(false, a0, false, b0a,
                                                             (short)0, c0a, false, false);
                for (int r = 0; r < 8; ++r)
                    h0s[wave][mbase + r][lh] = (_Float16)lrelu(c0a[r]);
            }
            {
                v16h b0b = *(const v16h*)sB0[t][1][lane];
                v8f c0b = {};
                c0b = __builtin_amdgcn_wmma_f32_16x16x32_f16(false, a0, false, b0b,
                                                             (short)0, c0b, false, false);
                for (int r = 0; r < 8; ++r)
                    h0s[wave][mbase + r][16 + lh] = (_Float16)lrelu(c0b[r]);
            }
            __builtin_amdgcn_wave_barrier();
            V16U a1;
            a1.h[0] = *(const v8h*)&h0s[wave][lh][kb];
            a1.h[1] = *(const v8h*)&h0s[wave][lh][kb + 8];
            __builtin_amdgcn_wave_barrier();

            bool sel[8];
            for (int r = 0; r < 8; ++r) sel[r] = (tsel[r] == t);

            for (int j = 0; j < 4; ++j) {
                v16h b1f = *(const v16h*)sB1[t][j][lane];
                v8f c1 = {};
                c1 = __builtin_amdgcn_wmma_f32_16x16x32_f16(false, a1.v, false, b1f,
                                                            (short)0, c1, false, false);
                float bbv = sb1[t][j * 16 + lh];
                for (int r = 0; r < 8; ++r) {
                    float v = lrelu(c1[r] + bbv);
                    if (t == 0) gacc[j][r] = sel[r] ? v : 0.0f;
                    else if (sel[r]) gacc[j][r] = v;
                }
            }
        }
        for (int r = 0; r < 8; ++r) {
            int m = mbase + r;
            float xr0 = xs[wave][m][0], xr1 = xs[wave][m][1], xr2 = xs[wave][m][2];
            for (int j = 0; j < 4; ++j) {
                float gv = gacc[j][r];
                dacc[j][0] += gv * xr0;
                dacc[j][1] += gv * xr1;
                dacc[j][2] += gv * xr2;
            }
        }
    }
    for (int j = 0; j < 4; ++j)
        for (int c = 0; c < 3; ++c)
            unsafeAtomicAdd(&dpart[(j * 16 + lh) * 3 + c], dacc[j][c]);
}

// ---------------------------------------------------------------------------
// Fitting net  d[192] -> 256 -> 256 -> 3
// ---------------------------------------------------------------------------
__global__ void __launch_bounds__(256)
k_fit(const float* __restrict__ dpart,
      const float* __restrict__ Wf1, const float* __restrict__ bf1,
      const float* __restrict__ Wf2, const float* __restrict__ bf2,
      const float* __restrict__ Wo,  const float* __restrict__ bo,
      float* __restrict__ out) {
    __shared__ float sd[192];
    __shared__ float sh1[256];
    __shared__ float sh2[256];
    int i = threadIdx.x;
    if (i < 192) sd[i] = dpart[i];
    __syncthreads();
    {
        float acc = bf1[i];
        #pragma unroll 8
        for (int k = 0; k < 192; ++k) acc += sd[k] * Wf1[k * 256 + i];
        sh1[i] = lrelu(acc);
    }
    __syncthreads();
    {
        float acc = bf2[i];
        #pragma unroll 8
        for (int k = 0; k < 256; ++k) acc += sh1[k] * Wf2[k * 256 + i];
        sh2[i] = lrelu(acc);
    }
    __syncthreads();
    if (i < 3) {
        float acc = bo[i];
        #pragma unroll 8
        for (int k = 0; k < 256; ++k) acc += sh2[k] * Wo[k * 3 + i];
        out[i] = acc;
    }
}

// ---------------------------------------------------------------------------
extern "C" void kernel_launch(void* const* d_in, const int* in_sizes, int n_in,
                              void* d_out, int out_size, void* d_ws, size_t ws_size,
                              hipStream_t stream) {
    (void)n_in; (void)out_size;
    const float* x     = (const float*)d_in[0];
    const int*   al    = (const int*)  d_in[1];
    const float* W0    = (const float*)d_in[2];
    const float* b0    = (const float*)d_in[3];
    const float* W1    = (const float*)d_in[4];
    const float* b1    = (const float*)d_in[5];
    const float* Wf1   = (const float*)d_in[6];
    const float* bf1   = (const float*)d_in[7];
    const float* Wf2   = (const float*)d_in[8];
    const float* bf2   = (const float*)d_in[9];
    const float* Wo    = (const float*)d_in[10];
    const float* bo    = (const float*)d_in[11];
    float* out = (float*)d_out;

    char*  ws      = (char*)d_ws;
    float* dpart   = (float*)ws;
    int*   cnt     = (int*)(ws + 768);
    int*   cur     = (int*)(ws + 784);
    int*   off     = (int*)(ws + 800);
    float* xsorted = (float*)(ws + 1024);

    int n  = in_sizes[0] / 3;
    int TT = (n + 15) >> 4;
    int blocks = (TT + NWAVES - 1) / NWAVES;
    if (blocks > 1024) blocks = 1024;
    if (blocks < 1) blocks = 1;

    size_t need = 1024 + (size_t)3 * (size_t)(n + 64) * sizeof(float);

    k_init<<<1, 256, 0, stream>>>(dpart, cnt);
    if (ws_size >= need) {
        int sblocks = (n + 255) / 256;
        if (sblocks > 1024) sblocks = 1024;
        if (sblocks < 1) sblocks = 1;
        k_hist<<<sblocks, 256, 0, stream>>>(al, cnt, n);
        k_offs<<<1, 64, 0, stream>>>(cnt, cur, off, xsorted);
        k_scatter<<<sblocks, 256, 0, stream>>>(x, al, cur, xsorted, n);
        k_embed_sorted<<<blocks, 256, 0, stream>>>(xsorted, W0, b0, W1, b1, off, dpart);
    } else {
        k_embed_mixed<<<blocks, 256, 0, stream>>>(x, al, W0, b0, W1, b1, dpart, n);
    }
    k_fit<<<1, 256, 0, stream>>>(dpart, Wf1, bf1, Wf2, bf2, Wo, bo, out);
}
